// inter_module_30116310680038
// MI455X (gfx1250) — matrix operand order repaired and checked by reading, not compiled, not run on hardware
//
#include <hip/hip_runtime.h>

// ---------------------------------------------------------------------------
// CDNA5 (gfx1250) implicit-GEMM 3x3 conv using v_wmma_f32_16x16x32_bf16.
// Dual-branch residual block (rgb/hsi), 8 fused conv launches.
// Wave tiling: 8 wave32 = 2 M-groups x 4 N-groups; each wave 32x32 outputs
// via 4 accumulators -> 1 ds_load_b128 + 1 global_load_b128 per WMMA.
// ---------------------------------------------------------------------------

typedef __attribute__((ext_vector_type(16))) __bf16 v16bf;
typedef __attribute__((ext_vector_type(8)))  float  v8f;

#define NBATCH 8
#define CCH    64
#define HH     256
#define WW     256
#define HWSZ   (HH * WW)
#define XT     128         // output pixels per workgroup (one row strip)
#define LDSX   (XT + 2)    // 130: padded x extent in LDS
#define WFRAG  36864       // packed bf16 weights per conv: 4 mb * 9 taps * 2 cb * 512

__device__ __forceinline__ unsigned short f2bf(float f) {
  unsigned int u = __float_as_uint(f);
  u += 0x7FFFu + ((u >> 16) & 1u);      // round-to-nearest-even
  return (unsigned short)(u >> 16);
}
__device__ __forceinline__ float bf2f(unsigned short s) {
  return __uint_as_float(((unsigned int)s) << 16);
}

// ---- fp32 -> bf16 tensor conversion (vectorized x4) -----------------------
__global__ void cvt_f32_bf16(const float* __restrict__ src,
                             unsigned short* __restrict__ dst, int n4) {
  int i = blockIdx.x * blockDim.x + threadIdx.x;
  if (i >= n4) return;
  float4 f = reinterpret_cast<const float4*>(src)[i];
  ushort4 o;
  o.x = f2bf(f.x); o.y = f2bf(f.y); o.z = f2bf(f.z); o.w = f2bf(f.w);
  reinterpret_cast<ushort4*>(dst)[i] = o;
}

// ---- pack conv weights [64,64,3,3] fp32 into WMMA A-fragment bf16 layout --
// Fragment (mb, t, cb) is 512 bf16 = 32 lanes x 16 elems, per ISA 16-bit
// A-matrix 16x32 layout: lane<16 holds K {0..7,16..23}, lane>=16 holds
// K {8..15,24..31}; VGPR v holds a K pair (2 halves).
__global__ void pack_weights(const float* __restrict__ w,
                             unsigned short* __restrict__ dst) {
  int idx = blockIdx.x * blockDim.x + threadIdx.x;
  if (idx >= 4 * 9 * 2 * 512) return;
  int j    = idx & 15;          // element within lane
  int lane = (idx >> 4) & 31;
  int frag = idx >> 9;          // 0..71
  int cb   = frag & 1;          // 32-channel K block
  int t    = (frag >> 1) % 9;   // tap
  int mb   = frag / 18;         // 16-channel M block
  int v = j >> 1, h = j & 1;
  int base8 = (lane < 16) ? 0 : 8;
  int k = (v < 4) ? (base8 + 2 * v + h) : (16 + base8 + 2 * (v - 4) + h);
  int o  = mb * 16 + (lane & 15);
  int c  = cb * 32 + k;
  int ky = t / 3, kx = t % 3;
  dst[idx] = f2bf(w[((o * CCH + c) * 3 + ky) * 3 + kx]);
}

// ---- cat_rgb: dst[:, 32+c, :, :] = src[:, c, :, :], c in 0..31 ------------
__global__ void concat_copy(const unsigned short* __restrict__ src,
                            unsigned short* __restrict__ dst, int nq) {
  int i = blockIdx.x * blockDim.x + threadIdx.x;
  if (i >= nq) return;
  int p4 = i & 16383;          // HWSZ/4
  int c  = (i >> 14) & 31;
  int nb = i >> 19;
  const uint2* s = reinterpret_cast<const uint2*>(src + (size_t)(nb * CCH + c) * HWSZ) + p4;
  uint2*       d = reinterpret_cast<uint2*>(dst + (size_t)(nb * CCH + 32 + c) * HWSZ) + p4;
  *d = *s;
}

// ---- fused conv3x3 (implicit GEMM, bf16 WMMA, f32 accumulate) -------------
// epilogue: +bias, optional bf16 residual, optional f32 residual, optional
// relu; writes bf16 and/or f32 output.
__global__ __launch_bounds__(256)
void conv3x3_wmma(const unsigned short* __restrict__ in,      // bf16 NCHW
                  const unsigned short* __restrict__ wpack,   // packed A frags
                  const float* __restrict__ bias,
                  const unsigned short* __restrict__ resid_bf,
                  const float* __restrict__ resid_f32,
                  unsigned short* __restrict__ out_bf,
                  float* __restrict__ out_f32,
                  int do_relu) {
  // LDS tile: [row 0..2][x 0..129][channel-pair 0..31] as 32-bit words
  // ({c=2p, c=2p+1} bf16) so one K-pair is one ds_load_b32; two runs of 4 consecutive
  // pair-words per fragment merge into ds_load_b128 pairs.
  __shared__ unsigned int smem_u[3 * LDSX * (CCH / 2)];   // 49,920 B

  const int x0  = blockIdx.x * XT;
  const int y   = blockIdx.y;
  const int nb  = blockIdx.z;
  const int tid = threadIdx.x;

  __builtin_prefetch(wpack, 0, 3);  // pull weight fragments toward the WGP

  // ---- stage zero-padded input tile into LDS (x fastest for coalescing) --
  // One 32-bit LDS word per iteration: two guarded 16-bit loads from the two
  // channel planes of the pair.
  for (int idx = tid; idx < 3 * LDSX * (CCH / 2); idx += 256) {
    int x = idx % LDSX;
    int r = (idx / LDSX) % 3;
    int p = idx / (LDSX * 3);
    int gy = y + r - 1;
    int gx = x0 + x - 1;
    unsigned int w = 0;
    if (gy >= 0 && gy < HH && gx >= 0 && gx < WW) {
      const size_t base = (size_t)(nb * CCH + 2 * p) * HWSZ + gy * WW + gx;
      unsigned int lo = in[base];
      unsigned int hi = in[base + HWSZ];
      w = lo | (hi << 16);
    }
    smem_u[(r * LDSX + x) * (CCH / 2) + p] = w;
  }
  __syncthreads();

  const int lane = tid & 31;
  const int wv   = tid >> 5;        // 8 wave32 per workgroup
  const int mg   = wv & 1;          // M group: output channels mg*32..+31
  const int xw   = (wv >> 1) * 32;  // N group: 32 pixels (two 16-px tiles)
  const int nloc = lane & 15;
  const int kb16 = (lane < 16) ? 0 : 16;   // B-fragment K base per ISA layout

  v8f acc00 = {};   // (m-tile 0, n-tile 0)
  v8f acc01 = {};   // (m-tile 0, n-tile 1)
  v8f acc10 = {};   // (m-tile 1, n-tile 0)
  v8f acc11 = {};   // (m-tile 1, n-tile 1)

  const unsigned short* wbase0 = wpack + (size_t)(mg * 2 + 0) * 9 * 2 * 512;
  const unsigned short* wbase1 = wpack + (size_t)(mg * 2 + 1) * 9 * 2 * 512;

  for (int t = 0; t < 9; ++t) {
    const int ky = t / 3, kx = t % 3;
    const int row0 = (ky * LDSX + (xw + nloc + kx)) * (CCH / 2);
    const int row1 = (ky * LDSX + (xw + 16 + nloc + kx)) * (CCH / 2);
    #pragma unroll
    for (int cb = 0; cb < 2; ++cb) {
      // A fragments: 16 bf16 per lane, pre-packed -> 2x global_load_b128 each
      const size_t foff = (size_t)(t * 2 + cb) * 512 + lane * 16;
      v16bf a0 = *reinterpret_cast<const v16bf*>(wbase0 + foff);
      v16bf a1 = *reinterpret_cast<const v16bf*>(wbase1 + foff);
      // B fragments: 8 channel-pair words per lane from LDS
      union { unsigned int u[8]; v16bf v; } b0, b1;
      const int pbase = cb * 16 + (kb16 >> 2);
      #pragma unroll
      for (int vv = 0; vv < 8; ++vv) {
        b0.u[vv] = smem_u[row0 + pbase + vv + (vv & 4)];
        b1.u[vv] = smem_u[row1 + pbase + vv + (vv & 4)];
      }
      acc00 = __builtin_amdgcn_wmma_f32_16x16x32_bf16(
          false, a0, false, b0.v, (short)0, acc00, false, false);
      acc01 = __builtin_amdgcn_wmma_f32_16x16x32_bf16(
          false, a0, false, b1.v, (short)0, acc01, false, false);
      acc10 = __builtin_amdgcn_wmma_f32_16x16x32_bf16(
          false, a1, false, b0.v, (short)0, acc10, false, false);
      acc11 = __builtin_amdgcn_wmma_f32_16x16x32_bf16(
          false, a1, false, b1.v, (short)0, acc11, false, false);
    }
  }

  // ---- epilogue: C/D layout VGPR v -> M = v + (lane<16?0:8), N = lane%16 --
  const int mhi = (lane < 16) ? 0 : 8;
  #pragma unroll
  for (int mt = 0; mt < 2; ++mt) {
    const int obase = mg * 32 + mt * 16;
    #pragma unroll
    for (int nt = 0; nt < 2; ++nt) {
      const v8f& acc = (mt == 0) ? (nt == 0 ? acc00 : acc01)
                                 : (nt == 0 ? acc10 : acc11);
      const int X = x0 + xw + nt * 16 + nloc;
      #pragma unroll
      for (int vv = 0; vv < 8; ++vv) {
        const int o  = obase + vv + mhi;
        const size_t idx = (size_t)(nb * CCH + o) * HWSZ + y * WW + X;
        float val = acc[vv] + bias[o];
        if (resid_bf)  val += bf2f(resid_bf[idx]);
        if (resid_f32) val += resid_f32[idx];
        if (do_relu)   val = fmaxf(val, 0.f);
        if (out_bf)    out_bf[idx]  = f2bf(val);
        if (out_f32)   out_f32[idx] = val;
      }
    }
  }
}

// ---------------------------------------------------------------------------
extern "C" void kernel_launch(void* const* d_in, const int* in_sizes, int n_in,
                              void* d_out, int out_size, void* d_ws, size_t ws_size,
                              hipStream_t stream) {
  (void)in_sizes; (void)n_in; (void)out_size; (void)ws_size;
  const float* rgb = (const float*)d_in[0];
  const float* hsi = (const float*)d_in[1];
  const size_t ACT = (size_t)NBATCH * CCH * HWSZ;   // 33,554,432 elements

  // workspace: packed weights + 3 bf16 activation buffers
  unsigned short* wp = (unsigned short*)d_ws;                       // 8*WFRAG shorts
  unsigned short* B2 = (unsigned short*)((char*)d_ws + (1u << 20)); // tmp
  unsigned short* B3 = B2 + ACT;                                    // out_rgb / cat_rgb
  unsigned short* B4 = B3 + ACT;                                    // out_hsi (==cat_hsi)
  // B0/B1 (bf16 copies of inputs) live in d_out; both dead before the final
  // f32 stores (uC2/uD2) overwrite those byte ranges.
  unsigned short* B0 = (unsigned short*)d_out;        // bytes [0, 67MB)
  unsigned short* B1 = B0 + ACT;                      // bytes [67MB, 134MB)

  float* out_rgb = (float*)d_out;
  float* out_hsi = out_rgb + ACT;

  const dim3 cgrid(WW / XT, HH, NBATCH);
  const dim3 cblk(256);

  // 1) inputs -> bf16
  const int n4 = (int)(ACT / 4);
  cvt_f32_bf16<<<dim3((n4 + 255) / 256), cblk, 0, stream>>>(rgb, B0, n4);
  cvt_f32_bf16<<<dim3((n4 + 255) / 256), cblk, 0, stream>>>(hsi, B1, n4);

  // 2) pack 8 conv weights (order: uA1,uA2,uB1,uB2,uC1,uC2,uD1,uD2)
  for (int i = 0; i < 8; ++i)
    pack_weights<<<dim3(144), cblk, 0, stream>>>(
        (const float*)d_in[2 + 2 * i], wp + (size_t)i * WFRAG);

  // 3) uA on rgb: out_rgb_pre = relu(conv2(relu(conv1(rgb))) + rgb)
  conv3x3_wmma<<<cgrid, cblk, 0, stream>>>(B0, wp + 0 * WFRAG, (const float*)d_in[3],
                                           nullptr, nullptr, B2, nullptr, 1);
  conv3x3_wmma<<<cgrid, cblk, 0, stream>>>(B2, wp + 1 * WFRAG, (const float*)d_in[5],
                                           B0, nullptr, B3, nullptr, 1);
  // 4) uB on hsi
  conv3x3_wmma<<<cgrid, cblk, 0, stream>>>(B1, wp + 2 * WFRAG, (const float*)d_in[7],
                                           nullptr, nullptr, B2, nullptr, 1);
  conv3x3_wmma<<<cgrid, cblk, 0, stream>>>(B2, wp + 3 * WFRAG, (const float*)d_in[9],
                                           B1, nullptr, B4, nullptr, 1);

  // 5) cat_rgb = [out_rgb[:, :32], out_hsi[:, :32]]  (cat_hsi == out_hsi)
  const int nq = (int)(NBATCH * 32 * HWSZ / 4);
  concat_copy<<<dim3(nq / 256), cblk, 0, stream>>>(B4, B3, nq);

  // 6) uC on cat_rgb: final_rgb = conv2(relu(conv1(cat))) + b + cat + rgb
  conv3x3_wmma<<<cgrid, cblk, 0, stream>>>(B3, wp + 4 * WFRAG, (const float*)d_in[11],
                                           nullptr, nullptr, B2, nullptr, 1);
  conv3x3_wmma<<<cgrid, cblk, 0, stream>>>(B2, wp + 5 * WFRAG, (const float*)d_in[13],
                                           B3, rgb, nullptr, out_rgb, 0);

  // 7) uD on out_hsi: final_hsi = conv2(relu(conv1(out_hsi))) + b + out_hsi + hsi
  conv3x3_wmma<<<cgrid, cblk, 0, stream>>>(B4, wp + 6 * WFRAG, (const float*)d_in[15],
                                           nullptr, nullptr, B2, nullptr, 1);
  conv3x3_wmma<<<cgrid, cblk, 0, stream>>>(B2, wp + 7 * WFRAG, (const float*)d_in[17],
                                           B4, hsi, nullptr, out_hsi, 0);
}
